// TimeModulation_15144054685708
// MI455X (gfx1250) — matrix operand order repaired and checked by reading, not compiled, not run on hardware
//
#include <hip/hip_runtime.h>

typedef __attribute__((ext_vector_type(16))) _Float16     v16h;
typedef __attribute__((ext_vector_type(8)))  float        v8f;
typedef __attribute__((ext_vector_type(4)))  unsigned int u32x4;
typedef __attribute__((ext_vector_type(8)))  int          i32x8;
typedef __attribute__((ext_vector_type(4)))  int          i32x4;

#define NB 6
#define NH 32
#define EE 128
#define NTOK (8 * 2048)
#define NTILES (NTOK / 16)
// fragment-ready f16 coefficient buffer:
// frag id = ((band*2 + g)*2 + kfrag)*8 + etile   (192 fragments)
// per fragment: 32 lanes x 16 halfs (32 bytes per lane, lane-contiguous)
#define NFRAGS (NB * 2 * 2 * 8)
#define FRAG_STRIDE 1024                 // 32 lanes * 32 B
#define WS_BYTES (NFRAGS * FRAG_STRIDE)  // 196608 = 192 KB
#define WS_DWORDS (WS_BYTES / 4)         // 49152 (fits tile_dim0's 16 bits)

#if defined(__HIP_DEVICE_COMPILE__) &&                              \
    __has_builtin(__builtin_amdgcn_tensor_load_to_lds) &&           \
    __has_builtin(__builtin_amdgcn_s_wait_tensorcnt)
#define USE_TDM 1
#else
#define USE_TDM 0
#endif

// --------------------------------------------------------------------------
// Prep: reorder alpha/beta (NB,H,E) f32 -> f16 WMMA-B fragment layout in ws.
// B k order used here (16-bit, 32x16; contiguous, same as A): lane%16 = column N, lanes 0-15 hold
// K=0..15 (2 per VGPR), lanes 16-31 hold K=16..31.  K = 2h + (0:sin/1:cos).
// --------------------------------------------------------------------------
__global__ void __launch_bounds__(256)
prep_coeff_kernel(const float* __restrict__ a1, const float* __restrict__ b1,
                  const float* __restrict__ a2, const float* __restrict__ b2,
                  _Float16* __restrict__ wsf) {
  int tid = blockIdx.x * blockDim.x + threadIdx.x;
  if (tid >= NFRAGS * 32 * 16) return;
  int i     = tid & 15;
  int lane  = (tid >> 4) & 31;
  int etile = (tid >> 9) & 7;
  int kfrag = (tid >> 12) & 1;
  int g     = (tid >> 13) & 1;
  int band  = tid >> 14;
  int K = i + ((lane >= 16) ? 16 : 0) + kfrag * 32;  // 0..63
  int h = K >> 1;                                    // harmonic index 0..31
  int p = K & 1;                                     // 0 = sin/alpha, 1 = cos/beta
  int e = etile * 16 + (lane & 15);
  const float* src = (g == 0) ? (p ? b1 : a1) : (p ? b2 : a2);
  wsf[tid] = (_Float16)src[(band * NH + h) * EE + e];
}

// --------------------------------------------------------------------------
// Main kernel.
// --------------------------------------------------------------------------
__global__ void __launch_bounds__(256)
TimeModulation_15144054685708_kernel(const float* __restrict__ flux,
                                     const float* __restrict__ time_,
                                     const int*   __restrict__ band_idx,
                                     const float* __restrict__ W,
                                     const float* __restrict__ bvec,
                                     const _Float16* __restrict__ wsf,
                                     float* __restrict__ out) {
  extern __shared__ unsigned char smem[];

  // ---- Stage the 192 KB fragment bank into LDS ----
#if USE_TDM
  if (threadIdx.x < 32) {  // wave 0 issues one TDM DMA (EXEC ignored by TDM)
    typedef __attribute__((address_space(3))) unsigned char lds_uchar;
    unsigned lds_base = (unsigned)(lds_uchar*)smem;          // LDS byte offset
    unsigned long long ga = (unsigned long long)wsf;         // global byte addr

    // D# group 0: count=1 (valid user desc), lds_addr, global_addr, type=2
    u32x4 g0;
    g0[0] = 1u;
    g0[1] = lds_base;
    g0[2] = (unsigned)(ga & 0xFFFFFFFFu);
    g0[3] = (unsigned)((ga >> 32) & 0x01FFFFFFu) | 0x80000000u;  // type=2

    // D# group 1: data_size=4B; 1-D tile: tensor_dim0 = tile_dim0 = 49152 dw
    i32x8 g1 = {};
    g1[0] = (int)(2u << 16);                          // data_size = 4 bytes
    g1[1] = (int)((WS_DWORDS & 0xFFFFu) << 16);       // tensor_dim0[15:0]
    g1[2] = (int)((WS_DWORDS >> 16) & 0xFFFFu) |      // tensor_dim0[31:16]
            (int)(1u << 16);                          // tensor_dim1 = 1
    g1[3] = (int)((WS_DWORDS & 0xFFFFu) << 16);       // tile_dim0
    g1[4] = 0;                                        // tile_dim1/2 unused
    g1[5] = (int)WS_DWORDS;                           // tensor_dim0_stride lo
    g1[6] = 0;
    g1[7] = 0;

    i32x4 gz = {};
#if __clang_major__ >= 23
    i32x8 gz8 = {};
    __builtin_amdgcn_tensor_load_to_lds(g0, g1, gz, gz, gz8, 0);
#else
    __builtin_amdgcn_tensor_load_to_lds(g0, g1, gz, gz, 0);
#endif
    __builtin_amdgcn_s_wait_tensorcnt(0);
  }
#else
  {
    const uint4* gsrc = (const uint4*)wsf;
    uint4*       sdst = (uint4*)smem;
    const int n16 = WS_BYTES / 16;  // 12288
    for (int idx = threadIdx.x; idx < n16; idx += blockDim.x)
      sdst[idx] = gsrc[idx];
  }
#endif
  __syncthreads();

  const int lane = threadIdx.x & 31;
  const int wave = threadIdx.x >> 5;
  const int hi   = lane >> 4;       // half-wave select
  const int mrow = lane & 15;       // A row / C column role of this lane

  for (int tile = blockIdx.x * 8 + wave; tile < NTILES; tile += gridDim.x * 8) {
    const int t0 = tile * 16;
    const int t  = t0 + mrow;
    const float tn    = time_[t] * (1.0f / 1500.0f);
    const int   bandm = band_idx[t];

    // ---- Build full A fragments (16x32 f16 each; two K-halves of K=64) ----
    // A k order used here (same as B): local K = i + (hi ? 16 : 0)
    v16h afull0, afull1;
#pragma unroll
    for (int i = 0; i < 16; ++i) {
      const int Ka = i + (hi ? 16 : 0);
      const int h  = Ka >> 1;
      const int p  = Ka & 1;
      const float ang0 = 6.28318530718f * tn * (float)(h + 1);
      const float ang1 = 6.28318530718f * tn * (float)(h + 17);
      afull0[i] = (_Float16)(p ? __cosf(ang0) : __sinf(ang0));
      afull1[i] = (_Float16)(p ? __cosf(ang1) : __sinf(ang1));
    }

    // ---- Per-band zero-masked A copies (branchless, EXEC stays full) ----
    v16h am0[NB], am1[NB];
#pragma unroll
    for (int bnd = 0; bnd < NB; ++bnd) {
      const _Float16 mk = (bandm == bnd) ? (_Float16)1.0f : (_Float16)0.0f;
      am0[bnd] = afull0 * mk;
      am1[bnd] = afull1 * mk;
    }

    // ---- Preload flux rows for the fused epilogue (C rows = v + hi*8) ----
    float fx0[8], fx1[8];
#pragma unroll
    for (int v = 0; v < 8; ++v) {
      const int row = t0 + v + hi * 8;
      fx0[v] = flux[row * 2 + 0];
      fx1[v] = flux[row * 2 + 1];
    }

    for (int etile = 0; etile < 8; ++etile) {
      auto ldfrag = [&](int bnd, int g, int kfrag) -> v16h {
        const int frag = ((bnd * 2 + g) * 2 + kfrag) * 8 + etile;
        const uint4* p = (const uint4*)(smem + frag * FRAG_STRIDE + lane * 32);
        union { uint4 u[2]; v16h v; } bb;
        bb.u[0] = p[0];
        bb.u[1] = p[1];
        return bb.v;
      };

      v8f acc1 = {};
      v8f acc2 = {};
      // Double-buffered B fragments: band bnd+1 loads issue while band bnd's
      // WMMAs execute, so s_wait_dscnt stalls overlap with XDL work.
      v16h f00 = ldfrag(0, 0, 0), f01 = ldfrag(0, 0, 1);
      v16h f10 = ldfrag(0, 1, 0), f11 = ldfrag(0, 1, 1);
#pragma unroll
      for (int bnd = 0; bnd < NB; ++bnd) {
        v16h n00, n01, n10, n11;
        if (bnd + 1 < NB) {
          n00 = ldfrag(bnd + 1, 0, 0);
          n01 = ldfrag(bnd + 1, 0, 1);
          n10 = ldfrag(bnd + 1, 1, 0);
          n11 = ldfrag(bnd + 1, 1, 1);
        }
        acc1 = __builtin_amdgcn_wmma_f32_16x16x32_f16(
            false, am0[bnd], false, f00, (short)0, acc1, false, false);
        acc2 = __builtin_amdgcn_wmma_f32_16x16x32_f16(
            false, am0[bnd], false, f10, (short)0, acc2, false, false);
        acc1 = __builtin_amdgcn_wmma_f32_16x16x32_f16(
            false, am1[bnd], false, f01, (short)0, acc1, false, false);
        acc2 = __builtin_amdgcn_wmma_f32_16x16x32_f16(
            false, am1[bnd], false, f11, (short)0, acc2, false, false);
        if (bnd + 1 < NB) { f00 = n00; f01 = n01; f10 = n10; f11 = n11; }
      }

      // ---- Fused epilogue: out = x*(1+g1) + g2, x = flux.W + b ----
      const int e  = etile * 16 + mrow;       // C-fragment column of this lane
      const float w0 = W[e * 2 + 0];
      const float w1 = W[e * 2 + 1];
      const float bb = bvec[e];
#pragma unroll
      for (int v = 0; v < 8; ++v) {
        const int row = t0 + v + hi * 8;      // C-fragment row M
        const float x = fx0[v] * w0 + fx1[v] * w1 + bb;
        out[(size_t)row * EE + e] = x * (1.0f + acc1[v]) + acc2[v];
      }
    }
  }
}

// --------------------------------------------------------------------------
extern "C" void kernel_launch(void* const* d_in, const int* in_sizes, int n_in,
                              void* d_out, int out_size, void* d_ws, size_t ws_size,
                              hipStream_t stream) {
  (void)in_sizes; (void)n_in; (void)out_size; (void)ws_size;
  const float* flux  = (const float*)d_in[0];
  const float* time_ = (const float*)d_in[1];
  const int*   band  = (const int*)d_in[2];
  const float* W     = (const float*)d_in[3];
  const float* b     = (const float*)d_in[4];
  const float* a1    = (const float*)d_in[5];
  const float* b1    = (const float*)d_in[6];
  const float* a2    = (const float*)d_in[7];
  const float* b2    = (const float*)d_in[8];
  float*       out   = (float*)d_out;
  _Float16*    wsf   = (_Float16*)d_ws;

  // 192 KB dynamic LDS (within gfx1250's 320 KB/WGP); opt in every call
  // (idempotent, deterministic, not a captured stream op).
  hipFuncSetAttribute((const void*)TimeModulation_15144054685708_kernel,
                      hipFuncAttributeMaxDynamicSharedMemorySize, WS_BYTES);

  // One thread per staged f16 coefficient: 192 frags * 32 lanes * 16 halves.
  prep_coeff_kernel<<<(NFRAGS * 32 * 16) / 256, 256, 0, stream>>>(a1, b1, a2, b2, wsf);

  // 1024 tiles of 16 tokens; 8 waves per block -> 128 blocks covers all.
  TimeModulation_15144054685708_kernel<<<128, 256, WS_BYTES, stream>>>(
      flux, time_, band, W, b, wsf, out);
}
